// DogBrain_46961172414994
// MI455X (gfx1250) — matrix operand order repaired and checked by reading, not compiled, not run on hardware
//
#include <hip/hip_runtime.h>

typedef __attribute__((ext_vector_type(16))) __bf16 v16bf;
typedef __attribute__((ext_vector_type(8)))  __bf16 v8bf;
typedef __attribute__((ext_vector_type(8)))  float  v8f;
typedef unsigned short u16;
typedef int v4i __attribute__((__vector_size__(16)));  // matches async-LDS builtin params

// ---- optional CDNA5 async global->LDS path (guarded; falls back to uint4 copy) ----
#if defined(__has_builtin)
#if __has_builtin(__builtin_amdgcn_global_load_async_to_lds_b128) && \
    __has_builtin(__builtin_amdgcn_s_wait_asynccnt)
#define USE_ASYNC_LDS 1
#endif
#endif

// ---------- bf16 helpers ----------
__device__ __forceinline__ float bf2f(u16 u) {
  unsigned v = ((unsigned)u) << 16;
  return __builtin_bit_cast(float, v);
}
__device__ __forceinline__ u16 f2bf(float f) {
  return __builtin_bit_cast(u16, (__bf16)f);
}

__device__ __forceinline__ v8f wmma_bf16(v16bf a, v16bf b, v8f c) {
  // v_wmma_f32_16x16x32_bf16  (neg_a, A, neg_b, B, c_mod, C, reuse_a, reuse_b)
  return __builtin_amdgcn_wmma_f32_16x16x32_bf16(false, a, false, b, (short)0, c,
                                                 false, false);
}

// A-tile (16x32 bf16, MxK) per ISA 7.12.2: lane m = lane&15, lanes>=16 shift K by +8,
// regs 0..7 = K[base..base+7], regs 8..15 = K[base+16..base+23]  -> two 16B vector loads.
// Requires: row base 16B-aligned, lda multiple of 8, columns [k0 .. k0+31] readable.
__device__ __forceinline__ v16bf load_a16(const u16* __restrict__ rowp, int k0g) {
  v8bf lo = *reinterpret_cast<const v8bf*>(rowp + k0g);
  v8bf hi = *reinterpret_cast<const v8bf*>(rowp + k0g + 16);
  return __builtin_shufflevector(lo, hi, 0, 1, 2, 3, 4, 5, 6, 7, 8, 9, 10, 11, 12,
                                 13, 14, 15);
}

// B-tile (32x16 bf16, KxN): lane -> N col (n0+lane&15), regs 0..7 = K[k0+8h..+7],
// regs 8..15 = K[k0+16+8h..+7] (h = lane>>4), matching the A table order.
// Requires padded weights: Kpad rows, Npad cols (mult 16), base 32B-aligned.
__device__ __forceinline__ v16bf load_b16(const u16* __restrict__ w, int ldb, int kk,
                                          int n0) {
  int l = threadIdx.x & 31;
  int kb = kk - l + ((l >> 4) << 3);
  const u16* p = w + (size_t)kb * ldb + n0 + (l & 15);
  v16bf r;
#pragma unroll
  for (int s = 0; s < 8; ++s) {
    r[s] = __builtin_bit_cast(__bf16, p[(size_t)s * ldb]);
    r[8 + s] = __builtin_bit_cast(__bf16, p[(size_t)(16 + s) * ldb]);
  }
  return r;
}

// ---------- prep kernels ----------
// padded f32 -> bf16 matrix conversion (zero fill outside RxC)
__global__ void k_cvt_pad(const float* __restrict__ s, int sld, int R, int C,
                          u16* __restrict__ d, int Rpad, int Cpad) {
  int i = blockIdx.x * 256 + threadIdx.x;
  if (i >= Rpad * Cpad) return;
  int r = i / Cpad, c = i % Cpad;
  d[i] = (r < R && c < C) ? f2bf(s[(size_t)r * sld + c]) : (u16)0;
}
__global__ void k_zero_u16(u16* __restrict__ d, int n) {
  int i = blockIdx.x * blockDim.x + threadIdx.x;
  if (i < n) d[i] = 0;
}

// ---------- input projection: inp = x_t @ W(32x143) + b -> INPBUF bf16[128][160] ----------
__global__ void k_inp_gemm(const float* __restrict__ X, const u16* __restrict__ W,
                           const float* __restrict__ bias, u16* __restrict__ INP) {
  int lane = threadIdx.x & 31, wave = threadIdx.x >> 5;
  int tile = blockIdx.x * 4 + wave;  // 72 tiles exactly (18 blocks x 4 waves)
  int mt = tile / 9, nt = tile % 9;
  int row0 = mt * 16, n0 = nt * 16;
  // A tile from f32 x (K=32 exactly): convert on the fly
  const float* p = X + (size_t)(row0 + (lane & 15)) * (512 * 32);
  int g8 = (lane >> 4) * 8;
  v16bf a;
#pragma unroll
  for (int v = 0; v < 8; ++v) {
    int kk = g8 + ((v < 4) ? 2 * v : 16 + 2 * (v - 4));
    a[2 * v] = (__bf16)p[kk];
    a[2 * v + 1] = (__bf16)p[kk + 1];
  }
  v16bf b = load_b16(W, 144, lane, n0);
  v8f acc = (v8f)(0.0f);
  acc = wmma_bf16(a, b, acc);
#pragma unroll
  for (int i = 0; i < 8; ++i) {
    int m = row0 + i + ((lane >> 4) << 3);
    int n = n0 + (lane & 15);
    if (n < 143) INP[m * 160 + n] = f2bf(acc[i] + bias[n]);
  }
}

// ---------- region LTC cell GEMM: hn = h + dt*exp(ls)*(tanh(xin@Wx + h@Wh + b) - h) ----------
__global__ void k_region_gemm(const u16* __restrict__ A1, int lda1, int K1,
                              const u16* __restrict__ A2, int K2,
                              const u16* __restrict__ Wx1, const u16* __restrict__ Wx2,
                              const u16* __restrict__ Wh, int ldb,
                              const u16* __restrict__ Hst, const float* __restrict__ bias,
                              const float* __restrict__ lstep, float dt, int h,
                              float* __restrict__ hnbuf) {
  int lane = threadIdx.x & 31, wave = threadIdx.x >> 5;
  int Ntiles = (h + 15) >> 4;
  int tiles = Ntiles << 3;  // 8 M-tiles (B=128)
  int tile = blockIdx.x * 4 + wave;
  if (tile >= tiles) return;  // wave-uniform exit; active WMMA waves keep EXEC all-ones
  int mt = tile / Ntiles, nt = tile % Ntiles;
  int row0 = mt * 16, n0 = nt * 16;
  int mrow = row0 + (lane & 15);
  int g8 = (lane >> 4) * 8;
  v8f acc = (v8f)(0.0f);
  for (int k0 = 0; k0 < K1; k0 += 32) {  // feed-forward term (zero-padded B rows)
    v16bf a = load_a16(A1 + (size_t)mrow * lda1, k0 + g8);
    v16bf b = load_b16(Wx1, ldb, k0 + lane, n0);
    acc = wmma_bf16(a, b, acc);
  }
  if (A2) {  // self-loop: xin = concat(xin, h); separately padded segment
    for (int k0 = 0; k0 < K2; k0 += 32) {
      v16bf a = load_a16(A2 + (size_t)mrow * 1024, k0 + g8);
      v16bf b = load_b16(Wx2, ldb, k0 + lane, n0);
      acc = wmma_bf16(a, b, acc);
    }
  }
  for (int k0 = 0; k0 < h; k0 += 32) {  // recurrent term h @ Wh
    v16bf a = load_a16(Hst + (size_t)mrow * 1024, k0 + g8);
    v16bf b = load_b16(Wh, ldb, k0 + lane, n0);
    acc = wmma_bf16(a, b, acc);
  }
#pragma unroll
  for (int i = 0; i < 8; ++i) {  // C layout: vgpr i -> M=i (+8 for upper half-wave)
    int m = row0 + i + ((lane >> 4) << 3);
    int n = n0 + (lane & 15);
    if (n < h) {
      float cand = tanhf(acc[i] + bias[n]);
      float hp = bf2f(Hst[m * 1024 + n]);
      float st = dt * __expf(lstep[n]);
      hnbuf[m * 144 + n] = hp + st * (cand - hp);
    }
  }
}

// ---------- per-row LayerNorm of a region (h <= 143), writes bf16 state ----------
__global__ void k_region_ln(const float* __restrict__ hn, const float* __restrict__ g,
                            const float* __restrict__ bb, u16* __restrict__ dst, int h) {
  int m = blockIdx.x, tid = threadIdx.x;  // 128 blocks x 256 threads
  __shared__ float s1[256], s2[256];
  float v = (tid < h) ? hn[m * 144 + tid] : 0.f;
  s1[tid] = v;
  s2[tid] = v * v;
  __syncthreads();
  for (int o = 128; o > 0; o >>= 1) {
    if (tid < o) { s1[tid] += s1[tid + o]; s2[tid] += s2[tid + o]; }
    __syncthreads();
  }
  float mean = s1[0] / (float)h;
  float var = s2[0] / (float)h - mean * mean;
  float rs = rsqrtf(var + 1e-5f);
  if (tid < h) dst[m * 1024 + tid] = f2bf(g[tid] * (v - mean) * rs + bb[tid]);
}

// ---------- axon link + caudate input (small: VALU) ----------
__global__ void k_link(const u16* __restrict__ C, const float* __restrict__ agw,
                       const float* __restrict__ agb, const float* __restrict__ apw,
                       const float* __restrict__ apb, u16* __restrict__ CD, int t) {
  int idx = blockIdx.x * blockDim.x + threadIdx.x;
  if (idx >= 128 * 81) return;
  int m = idx / 81, j = idx % 81;
  const u16* soc = C + m * 1024 + 732;  // social_cortex slice (prev step value)
  float dg = agb[j], dp = apb[j];
  for (int k = 0; k < 102; ++k) {
    float s = bf2f(soc[k]);
    dg += s * agw[k * 81 + j];
    dp += s * apw[k * 81 + j];
  }
  float link = (1.f / (1.f + __expf(-dg))) * tanhf(dp);
  float hip = bf2f(C[m * 1024 + 468 + j]);  // hippocampus (updated this step)
  CD[m * 96 + j] = f2bf(hip + ((t > 0) ? 0.2f : 0.f) * link);
}

// ---------- dual big GEMM: allr @ tgW and allr @ thW, A staged in LDS ----------
__global__ void k_gate_gemm(const u16* __restrict__ C, const u16* __restrict__ TGW,
                            const u16* __restrict__ THW, const float* __restrict__ tgb,
                            const float* __restrict__ thb, float* __restrict__ TG,
                            float* __restrict__ TH) {
  __shared__ u16 sA[16 * 1024];  // 32 KB: one M-tile of state, shared by 8 waves
  int tid = threadIdx.x;
  int lane = tid & 31, wave = tid >> 5;
  int mt = blockIdx.x >> 3;                   // 8 blocks per M-tile
  int nt = ((blockIdx.x & 7) << 3) + wave;    // 64 N-tiles
  int row0 = mt * 16, n0 = nt * 16;
  // stage 16 full rows of CONCAT (contiguous 32 KB)
  {
    const u16* srcp = C + (size_t)row0 * 1024;
#ifdef USE_ASYNC_LDS
    // async DMA into LDS: 2048 x 16B transfers, tracked on ASYNCcnt
#pragma unroll
    for (int j = 0; j < 8; ++j) {
      size_t eo = ((size_t)tid + (size_t)j * 256) * 8;  // u16 units, 8 u16 = 16B
      __builtin_amdgcn_global_load_async_to_lds_b128((v4i*)(srcp + eo),
                                                     (v4i*)(sA + eo), 0, 0);
    }
    __builtin_amdgcn_s_wait_asynccnt(0);
#else
    const uint4* src = reinterpret_cast<const uint4*>(srcp);
    uint4* dst = reinterpret_cast<uint4*>(sA);
#pragma unroll
    for (int j = 0; j < 8; ++j) dst[tid + j * 256] = src[tid + j * 256];
#endif
  }
  __syncthreads();
  const u16* arow = sA + (size_t)(lane & 15) * 1024;
  int g8 = (lane >> 4) * 8;
  v8f c1 = (v8f)(0.0f), c2 = (v8f)(0.0f);
  for (int k0 = 0; k0 < 1017; k0 += 32) {
    v16bf a = load_a16(arow, k0 + g8);  // ds_load path; cols 1017..1023 are zero
    v16bf b1 = load_b16(TGW, 1024, k0 + lane, n0);
    v16bf b2 = load_b16(THW, 1024, k0 + lane, n0);
    if (k0 + 32 < 1017) {
      __builtin_prefetch(TGW + (size_t)(k0 + 32) * 1024 + n0, 0, 1);
      __builtin_prefetch(THW + (size_t)(k0 + 32) * 1024 + n0, 0, 1);
    }
    c1 = wmma_bf16(a, b1, c1);
    c2 = wmma_bf16(a, b2, c2);
  }
#pragma unroll
  for (int i = 0; i < 8; ++i) {
    int m = row0 + i + ((lane >> 4) << 3);
    int n = n0 + (lane & 15);
    TG[m * 1024 + n] = c1[i] + tgb[n];
    TH[m * 1024 + n] = c2[i] + thb[n];
  }
}

// ---------- out = sigmoid(LN(tg)) * th -> hidden[:,t,:] ----------
__global__ void k_gate_out(const float* __restrict__ TG, const float* __restrict__ TH,
                           const float* __restrict__ g, const float* __restrict__ bb,
                           float* __restrict__ hid, int t) {
  int m = blockIdx.x, tid = threadIdx.x;  // 128 x 256
  __shared__ float s1[256], s2[256];
  const float* row = TG + m * 1024;
  float ls = 0.f, lq = 0.f;
  for (int j = tid; j < 1024; j += 256) { float v = row[j]; ls += v; lq += v * v; }
  s1[tid] = ls; s2[tid] = lq;
  __syncthreads();
  for (int o = 128; o > 0; o >>= 1) {
    if (tid < o) { s1[tid] += s1[tid + o]; s2[tid] += s2[tid + o]; }
    __syncthreads();
  }
  float mean = s1[0] / 1024.f;
  float rs = rsqrtf(s2[0] / 1024.f - mean * mean + 1e-5f);
  float* o = hid + ((size_t)m * 512 + t) * 1024;
  for (int j = tid; j < 1024; j += 256) {
    float v = row[j];
    float gate = 1.f / (1.f + __expf(-(g[j] * (v - mean) * rs + bb[j])));
    o[j] = gate * TH[m * 1024 + j];
  }
}

// ---------- per-row mean/var of hidden (65536 rows) ----------
__global__ void k_meanvar(const float* __restrict__ hid, float* __restrict__ MV) {
  int wave = threadIdx.x >> 5, lane = threadIdx.x & 31;
  int row = blockIdx.x * 8 + wave;
  const float* p = hid + (size_t)row * 1024;
  float s = 0.f, q = 0.f;
  for (int j = 0; j < 32; ++j) { float v = p[lane + 32 * j]; s += v; q += v * v; }
  for (int m = 16; m > 0; m >>= 1) { s += __shfl_xor(s, m, 32); q += __shfl_xor(q, m, 32); }
  if (lane == 0) {
    float mean = s / 1024.f;
    MV[2 * row] = mean;
    MV[2 * row + 1] = q / 1024.f - mean * mean;
  }
}

// ---------- predictions = LN_nsh(hidden[:,:-1]) @ nshW + b (WMMA, LN fused in A loader) ----------
__global__ void k_pred(const float* __restrict__ hid, const float* __restrict__ MV,
                       const float* __restrict__ lng, const float* __restrict__ lnb,
                       const u16* __restrict__ W, const float* __restrict__ bias,
                       float* __restrict__ out) {
  int lane = threadIdx.x & 31, wave = threadIdx.x >> 5;
  int tile = blockIdx.x * 8 + wave;  // 8176 tiles exactly (1022 blocks x 8 waves)
  int mt = tile >> 1, nt = tile & 1;
  int row0 = mt * 16, n0 = nt * 16;
  int r = row0 + (lane & 15);  // 65408 = 4088*16, always valid
  int bidx = r / 511, tt = r % 511;
  int hrow = bidx * 512 + tt;
  const float* p = hid + (size_t)hrow * 1024;
  float mean = MV[2 * hrow];
  float rs = rsqrtf(MV[2 * hrow + 1] + 1e-5f);
  int g8 = (lane >> 4) * 8;
  v8f acc = (v8f)(0.0f);
  for (int k0 = 0; k0 < 1024; k0 += 32) {
    v16bf a;
#pragma unroll
    for (int v = 0; v < 8; ++v) {
      int kk = k0 + g8 + ((v < 4) ? 2 * v : 16 + 2 * (v - 4));
      a[2 * v] = (__bf16)((p[kk] - mean) * rs * lng[kk] + lnb[kk]);
      a[2 * v + 1] = (__bf16)((p[kk + 1] - mean) * rs * lng[kk + 1] + lnb[kk + 1]);
    }
    v16bf b = load_b16(W, 32, k0 + lane, n0);
    acc = wmma_bf16(a, b, acc);
  }
#pragma unroll
  for (int i = 0; i < 8; ++i) {
    int m = row0 + i + ((lane >> 4) << 3);
    int n = n0 + (lane & 15);
    if (n < 32) out[(size_t)m * 32 + n] = acc[i] + bias[n];
  }
}

// ---------- anomaly (per-row dot) + surprise ----------
__global__ void k_anomsurp(const float* __restrict__ hid, const float* __restrict__ MV,
                           const float* __restrict__ g, const float* __restrict__ bb,
                           const float* __restrict__ w, const float* __restrict__ blin,
                           const float* __restrict__ pred, const float* __restrict__ X,
                           float* __restrict__ anom, float* __restrict__ surp) {
  int wave = threadIdx.x >> 5, lane = threadIdx.x & 31;
  int row = blockIdx.x * 8 + wave;  // 65536 rows
  int b = row >> 9, t = row & 511;
  const float* p = hid + (size_t)row * 1024;
  float mean = MV[2 * row];
  float rs = rsqrtf(MV[2 * row + 1] + 1e-5f);
  float s = 0.f;
  for (int j = 0; j < 32; ++j) {
    int k = lane + 32 * j;
    s += ((p[k] - mean) * rs * g[k] + bb[k]) * w[k];
  }
  for (int m = 16; m > 0; m >>= 1) s += __shfl_xor(s, m, 32);
  if (lane == 0) anom[row] = 1.f / (1.f + __expf(-(s + blin[0])));
  if (t < 511) {
    int pr = b * 511 + t;
    float d = pred[(size_t)pr * 32 + lane] - X[((size_t)b * 512 + t + 1) * 32 + lane];
    float q = d * d;
    for (int m = 16; m > 0; m >>= 1) q += __shfl_xor(q, m, 32);
    if (lane == 0) surp[pr] = q / 32.f;
  }
}

// ---------- properties head (tiny: VALU) ----------
__global__ void k_props(const float* __restrict__ hid, const float* __restrict__ MV,
                        const float* __restrict__ g, const float* __restrict__ bb,
                        const float* __restrict__ w1, const float* __restrict__ b1,
                        const float* __restrict__ w2, const float* __restrict__ b2,
                        float* __restrict__ out) {
  int b = blockIdx.x, tid = threadIdx.x;  // 128 x 256
  __shared__ float xr[1024];
  __shared__ float h2[512];
  int row = b * 512 + 511;
  const float* p = hid + (size_t)row * 1024;
  float mean = MV[2 * row];
  float rs = rsqrtf(MV[2 * row + 1] + 1e-5f);
  for (int k = tid; k < 1024; k += 256) xr[k] = (p[k] - mean) * rs * g[k] + bb[k];
  __syncthreads();
  for (int o = tid; o < 512; o += 256) {
    float d = b1[o];
    for (int k = 0; k < 1024; ++k) d += xr[k] * w1[(size_t)k * 512 + o];
    h2[o] = 0.5f * d * (1.f + erff(d * 0.70710678f));  // exact gelu
  }
  __syncthreads();
  if (tid < 2) {
    float d = b2[tid];
    for (int k = 0; k < 512; ++k) d += h2[k] * w2[k * 2 + tid];
    out[b * 2 + tid] = d;
  }
}

// =====================================================================
static inline size_t pad16h(size_t n) { return (n + 15) & ~(size_t)15; }
static inline size_t pad32h(size_t n) { return (n + 31) & ~(size_t)31; }

extern "C" void kernel_launch(void* const* d_in, const int* in_sizes, int n_in,
                              void* d_out, int out_size, void* d_ws, size_t ws_size,
                              hipStream_t stream) {
  (void)in_sizes; (void)n_in; (void)out_size; (void)ws_size;
  // region metadata (NAMES order)
  static const int R_H[12]   = {143, 81, 61, 61, 61, 61, 81, 102, 81, 102, 81, 102};
  static const int R_IN[12]  = {143, 143, 143, 61, 285, 61, 61, 81, 102, 102, 81, 81};
  static const int R_SELF[12]= {0, 0, 0, 0, 0, 0, 1, 0, 1, 1, 1, 0};
  static const int R_OFF[12] = {0, 143, 224, 285, 346, 407, 468, 549, 651, 732, 834, 915};
  static const float R_DT[12]= {2.f, 1.5f, 1.f, 1.f, 1.f, 1.5f, 1.f, 1.5f, 0.8f, 1.f, 1.f, 1.5f};
  static const int EXEC[12]  = {0, 1, 2, 4, 3, 5, 6, 7, 9, 8, 10, 11};  // reference order
  static const int IN_OFF[12]= {-1, -1, -1, 346, 0, 285, 407, -2, 732, 549, 651, 834};

  const float* X    = (const float*)d_in[0];
  const float* inpW = (const float*)d_in[1];
  const float* inpB = (const float*)d_in[2];
  auto RW = [&](int i, int j) { return (const float*)d_in[3 + 6 * i + j]; };
  const float* apW = (const float*)d_in[75]; const float* apB = (const float*)d_in[76];
  const float* agW = (const float*)d_in[77]; const float* agB = (const float*)d_in[78];
  const float* tgW = (const float*)d_in[79]; const float* tgB = (const float*)d_in[80];
  const float* tgLNg = (const float*)d_in[81]; const float* tgLNb = (const float*)d_in[82];
  const float* thW = (const float*)d_in[83]; const float* thB = (const float*)d_in[84];
  const float* nshLNg = (const float*)d_in[85]; const float* nshLNb = (const float*)d_in[86];
  const float* nshW = (const float*)d_in[87]; const float* nshB = (const float*)d_in[88];
  const float* ahLNg = (const float*)d_in[89]; const float* ahLNb = (const float*)d_in[90];
  const float* ahW = (const float*)d_in[91]; const float* ahB = (const float*)d_in[92];
  const float* phLNg = (const float*)d_in[93]; const float* phLNb = (const float*)d_in[94];
  const float* ph1W = (const float*)d_in[95]; const float* ph1B = (const float*)d_in[96];
  const float* ph2W = (const float*)d_in[97]; const float* ph2B = (const float*)d_in[98];

  // ---- workspace carve (all bf16 weight blocks padded: rows->mult32, cols->mult16) ----
  char* base = (char*)d_ws;
  u16* WB = (u16*)base;
  size_t cur = 0;
  auto alloc = [&](size_t r, size_t c) {
    size_t o = cur;
    cur = (cur + r * c + 15) & ~(size_t)15;  // keep every block 32B-aligned
    return o;
  };
  size_t inpo = alloc(32, 144);
  size_t wx1o[12], wx2o[12], who[12], ldw[12];
  for (int i = 0; i < 12; ++i) {
    ldw[i] = pad16h(R_H[i]);
    wx1o[i] = alloc(pad32h(R_IN[i]), ldw[i]);
    wx2o[i] = R_SELF[i] ? alloc(pad32h(R_H[i]), ldw[i]) : 0;
    who[i] = alloc(pad32h(R_H[i]), ldw[i]);
  }
  size_t tgo = alloc(1024, 1024);
  size_t tho = alloc(1024, 1024);
  size_t nsho = alloc(1024, 32);
  size_t ofs = (cur * 2 + 255) & ~(size_t)255;
  u16* CONCAT = (u16*)(base + ofs); ofs += (size_t)128 * 1024 * 2;
  u16* INP = (u16*)(base + ofs);    ofs += (size_t)128 * 160 * 2;
  u16* CD = (u16*)(base + ofs);     ofs += (size_t)128 * 96 * 2;
  ofs = (ofs + 255) & ~(size_t)255;
  float* HN = (float*)(base + ofs); ofs += (size_t)128 * 144 * 4;
  float* TG = (float*)(base + ofs); ofs += (size_t)128 * 1024 * 4;
  float* TH = (float*)(base + ofs); ofs += (size_t)128 * 1024 * 4;
  float* MV = (float*)(base + ofs); ofs += (size_t)65536 * 2 * 4;

  // ---- output carve (tuple return order) ----
  float* OUT = (float*)d_out;
  float* PRED = OUT;                                   // 128*511*32
  float* HID = OUT + (size_t)128 * 511 * 32;           // 128*512*1024
  float* ANOM = HID + (size_t)128 * 512 * 1024;        // 128*512
  float* SURP = ANOM + (size_t)128 * 512;              // 128*511
  float* PROPS = SURP + (size_t)128 * 511;             // 128*2

  // ---- prep: padded bf16 weight conversion + state zero ----
  auto cvt = [&](const float* s, int sld, int R, int C, size_t o, size_t Rp, size_t Cp) {
    int n = (int)(Rp * Cp);
    k_cvt_pad<<<(n + 255) / 256, 256, 0, stream>>>(s, sld, R, C, WB + o, (int)Rp, (int)Cp);
  };
  cvt(inpW, 143, 32, 143, inpo, 32, 144);
  for (int i = 0; i < 12; ++i) {
    int h = R_H[i], K1 = R_IN[i];
    cvt(RW(i, 0), h, K1, h, wx1o[i], pad32h(K1), ldw[i]);
    if (R_SELF[i])  // second segment of wx: rows K1..K1+h-1, padded separately
      cvt(RW(i, 0) + (size_t)K1 * h, h, h, h, wx2o[i], pad32h(h), ldw[i]);
    cvt(RW(i, 1), h, h, h, who[i], pad32h(h), ldw[i]);
  }
  cvt(tgW, 1024, 1017, 1024, tgo, 1024, 1024);
  cvt(thW, 1024, 1017, 1024, tho, 1024, 1024);
  cvt(nshW, 32, 1024, 32, nsho, 1024, 32);
  k_zero_u16<<<(128 * 1024 + 255) / 256, 256, 0, stream>>>(CONCAT, 128 * 1024);
  k_zero_u16<<<(128 * 160 + 255) / 256, 256, 0, stream>>>(INP, 128 * 160);
  k_zero_u16<<<(128 * 96 + 255) / 256, 256, 0, stream>>>(CD, 128 * 96);

  // ---- recurrent scan ----
  for (int t = 0; t < 512; ++t) {
    k_inp_gemm<<<18, 128, 0, stream>>>(X + (size_t)t * 32, WB + inpo, inpB, INP);
    for (int e = 0; e < 12; ++e) {
      int i = EXEC[e];
      if (i == 7)  // caudate input = hippocampus + 0.2*link(social_prev)
        k_link<<<(128 * 81 + 255) / 256, 256, 0, stream>>>(CONCAT, agW, agB, apW, apB, CD, t);
      const u16* A1; int lda1, K1;
      if (i < 3)       { A1 = INP; lda1 = 160; K1 = 143; }
      else if (i == 7) { A1 = CD;  lda1 = 96;  K1 = 81; }
      else             { A1 = CONCAT + IN_OFF[i]; lda1 = 1024; K1 = R_IN[i]; }
      const u16* A2 = R_SELF[i] ? (CONCAT + R_OFF[i]) : (const u16*)nullptr;
      int h = R_H[i];
      int Ntiles = (h + 15) / 16;
      int blocks = (8 * Ntiles + 3) / 4;
      k_region_gemm<<<blocks, 128, 0, stream>>>(
          A1, lda1, K1, A2, h, WB + wx1o[i], R_SELF[i] ? WB + wx2o[i] : nullptr,
          WB + who[i], (int)ldw[i], CONCAT + R_OFF[i], RW(i, 2), RW(i, 3), R_DT[i], h, HN);
      k_region_ln<<<128, 256, 0, stream>>>(HN, RW(i, 4), RW(i, 5), CONCAT + R_OFF[i], h);
    }
    k_gate_gemm<<<64, 256, 0, stream>>>(CONCAT, WB + tgo, WB + tho, tgB, thB, TG, TH);
    k_gate_out<<<128, 256, 0, stream>>>(TG, TH, tgLNg, tgLNb, HID, t);
  }

  // ---- heads ----
  k_meanvar<<<8192, 256, 0, stream>>>(HID, MV);
  k_pred<<<1022, 256, 0, stream>>>(HID, MV, nshLNg, nshLNb, WB + nsho, nshB, PRED);
  k_anomsurp<<<8192, 256, 0, stream>>>(HID, MV, ahLNg, ahLNb, ahW, ahB, PRED, X, ANOM, SURP);
  k_props<<<128, 256, 0, stream>>>(HID, MV, phLNg, phLNb, ph1W, ph1B, ph2W, ph2B, PROPS);
}
